// MegaCRN_53867479826913
// MI455X (gfx1250) — matrix operand order repaired and checked by reading, not compiled, not run on hardware
//
#include <hip/hip_runtime.h>
#include <hip/hip_bf16.h>
#include <math.h>

// ---------------------------------------------------------------------------
// MegaCRN forward on gfx1250 (MI455X).  All heavy GEMMs use
// v_wmma_f32_16x16x32_f16 (f16 in, f32 accumulate).  Every GEMM is arranged
// NT so both operands are loaded with contiguous 16B vector loads; each wave
// computes a 16x64 output strip (4 WMMAs per A load).  The k-loop is
// software-pipelined (depth 1): next iteration's 10 b128 loads are issued
// into fresh registers before this iteration's 4 WMMAs execute, so loads
// overlap matrix ops instead of serializing on s_wait_loadcnt 0.
// Scratch: ~165 MB in d_ws.
// ---------------------------------------------------------------------------

typedef _Float16 half_t;
typedef __attribute__((ext_vector_type(16))) _Float16 v16h;
typedef __attribute__((ext_vector_type(8)))  _Float16 h8;
typedef __attribute__((ext_vector_type(8)))  float    v8f;

#define DEVINL __device__ __forceinline__

// Problem constants
#define NN_  1000
#define NP   1024        // padded node count
#define BB   32
#define TT   12
#define NB   32000       // N*B valid rows
#define NBP  32768       // NP*B padded rows

// ---------------------------------------------------------------------------
// WMMA tile helpers (layouts per CDNA5 ISA 7.12.2, wave32)
// ---------------------------------------------------------------------------

// A-matrix 16x32 f16: lane = M(0..15) + K-half select; two 16B loads per lane.
DEVINL v16h wmma_load_a(const half_t* __restrict__ A, int lda) {
  const int lane = threadIdx.x & 31;
  const int m  = lane & 15;
  const int kb = (lane & 16) ? 8 : 0;
  const half_t* p = A + m * lda + kb;
  h8 lo = *(const h8*)(p);
  h8 hi = *(const h8*)(p + 16);
  v16h a;
#pragma unroll
  for (int j = 0; j < 8; ++j) { a[j] = lo[j]; a[8 + j] = hi[j]; }
  return a;
}

// B-matrix 32x16 from row-major Bt[N][K] (NT GEMM): contiguous 16B loads.
DEVINL v16h wmma_load_b_nt(const half_t* __restrict__ Bt, int ldb) {
  const int lane = threadIdx.x & 31;
  const int n  = lane & 15;
  const int kb = (lane & 16) ? 8 : 0;
  const half_t* p = Bt + n * ldb + kb;
  h8 lo = *(const h8*)(p);
  h8 hi = *(const h8*)(p + 16);
  v16h b;
#pragma unroll
  for (int j = 0; j < 8; ++j) { b[j] = lo[j]; b[8 + j] = hi[j]; }
  return b;
}

DEVINL v8f wmma_f16(v16h a, v16h b, v8f c) {
  return __builtin_amdgcn_wmma_f32_16x16x32_f16(false, a, false, b, (short)0, c,
                                                false, false);
}

// C/D 16x16 f32: lane = N(0..15) + M-half; VGPR j -> row mb+j.
DEVINL void wmma_store_f32(float* __restrict__ C, int ldc, v8f d) {
  const int lane = threadIdx.x & 31;
  const int n  = lane & 15;
  const int mb = (lane & 16) ? 8 : 0;
#pragma unroll
  for (int j = 0; j < 8; ++j) C[(mb + j) * ldc + n] = d[j];
}

// Software-pipelined 16x64-strip GEMM body.  K must be a multiple of 32, >= 32.
DEVINL void gemm4_pipeline(const half_t* __restrict__ Ap, int lda,
                           const half_t* __restrict__ Bp, int ldb,
                           int K, v8f acc[4]) {
  v16h a  = wmma_load_a(Ap, lda);
  v16h b0 = wmma_load_b_nt(Bp, ldb);
  v16h b1 = wmma_load_b_nt(Bp + (size_t)16 * ldb, ldb);
  v16h b2 = wmma_load_b_nt(Bp + (size_t)32 * ldb, ldb);
  v16h b3 = wmma_load_b_nt(Bp + (size_t)48 * ldb, ldb);
  for (int k0 = 32; k0 < K; k0 += 32) {
    // issue next iteration's loads into fresh registers first ...
    v16h an  = wmma_load_a(Ap + k0, lda);
    v16h bn0 = wmma_load_b_nt(Bp + k0, ldb);
    v16h bn1 = wmma_load_b_nt(Bp + (size_t)16 * ldb + k0, ldb);
    v16h bn2 = wmma_load_b_nt(Bp + (size_t)32 * ldb + k0, ldb);
    v16h bn3 = wmma_load_b_nt(Bp + (size_t)48 * ldb + k0, ldb);
    // ... then run this iteration's WMMAs on data loaded last iteration
    acc[0] = wmma_f16(a, b0, acc[0]);
    acc[1] = wmma_f16(a, b1, acc[1]);
    acc[2] = wmma_f16(a, b2, acc[2]);
    acc[3] = wmma_f16(a, b3, acc[3]);
    a = an; b0 = bn0; b1 = bn1; b2 = bn2; b3 = bn3;
  }
  acc[0] = wmma_f16(a, b0, acc[0]);
  acc[1] = wmma_f16(a, b1, acc[1]);
  acc[2] = wmma_f16(a, b2, acc[2]);
  acc[3] = wmma_f16(a, b3, acc[3]);
}

// ---------------------------------------------------------------------------
// GEMM kernels: 16x64 strip (4 x 16x16 tiles) per wave32, 4 waves per block.
// All NT: A[M x K] row-major, Bt[N x K] row-major (i.e. B = Bt^T).
// ---------------------------------------------------------------------------

// C(f32) = A * Bt^T      (ne @ ne^T scores)
__global__ void k_gemm4_nt_f32(const half_t* __restrict__ A,
                               const half_t* __restrict__ Bt,
                               float* __restrict__ C,
                               int Mt, int Nt, int K, int lda, int ldb, int ldc) {
  int wid = blockIdx.x * 4 + (threadIdx.x >> 5);
  int sn = Nt >> 2;
  if (wid >= Mt * sn) return;
  int tm = wid / sn, ts = wid % sn;
  v8f z = {};
  v8f acc[4] = {z, z, z, z};
  gemm4_pipeline(A + (size_t)tm * 16 * lda, lda,
                 Bt + (size_t)ts * 64 * ldb, ldb, K, acc);
  float* Cp = C + (size_t)tm * 16 * ldc + (size_t)ts * 64;
#pragma unroll
  for (int s = 0; s < 4; ++s) wmma_store_f32(Cp + 16 * s, ldc, acc[s]);
}

// C(f16) = A * Bt^T      (cheb!=0: C = 2*A*Bt^T - I)
__global__ void k_gemm4_nt_f16(const half_t* __restrict__ A,
                               const half_t* __restrict__ Bt,
                               half_t* __restrict__ C,
                               int Mt, int Nt, int K, int lda, int ldb, int ldc,
                               int cheb) {
  int wid = blockIdx.x * 4 + (threadIdx.x >> 5);
  int sn = Nt >> 2;
  if (wid >= Mt * sn) return;
  int tm = wid / sn, ts = wid % sn;
  v8f z = {};
  v8f acc[4] = {z, z, z, z};
  gemm4_pipeline(A + (size_t)tm * 16 * lda, lda,
                 Bt + (size_t)ts * 64 * ldb, ldb, K, acc);
  const int lane = threadIdx.x & 31;
  const int n  = lane & 15;
  const int mb = (lane & 16) ? 8 : 0;
  half_t* Cp = C + (size_t)tm * 16 * ldc + (size_t)ts * 64;
  const int gm = tm * 16 + mb;
#pragma unroll
  for (int s = 0; s < 4; ++s) {
    const int gn = ts * 64 + 16 * s + n;
#pragma unroll
    for (int j = 0; j < 8; ++j) {
      float v = acc[s][j];
      if (cheb) v = 2.0f * v - ((gm + j == gn) ? 1.0f : 0.0f);
      Cp[(mb + j) * ldc + 16 * s + n] = (half_t)v;
    }
  }
}

// C(f32) += A * Bt^T     (weight projection, accumulated over supports)
__global__ void k_gemm4_nt_accf32(const half_t* __restrict__ A,
                                  const half_t* __restrict__ Bt,
                                  float* __restrict__ C,
                                  int Mt, int Nt, int K, int lda, int ldb, int ldc) {
  int wid = blockIdx.x * 4 + (threadIdx.x >> 5);
  int sn = Nt >> 2;
  if (wid >= Mt * sn) return;
  int tm = wid / sn, ts = wid % sn;
  v8f z = {};
  v8f acc[4] = {z, z, z, z};
  gemm4_pipeline(A + (size_t)tm * 16 * lda, lda,
                 Bt + (size_t)ts * 64 * ldb, ldb, K, acc);
  const int lane = threadIdx.x & 31;
  const int n  = lane & 15;
  const int mb = (lane & 16) ? 8 : 0;
  float* Cp = C + (size_t)tm * 16 * ldc + (size_t)ts * 64;
#pragma unroll
  for (int s = 0; s < 4; ++s) {
#pragma unroll
    for (int j = 0; j < 8; ++j) {
      float* p = Cp + (mb + j) * ldc + 16 * s + n;
      *p = *p + acc[s][j];
    }
  }
}

// ---------------------------------------------------------------------------
// Small / elementwise kernels
// ---------------------------------------------------------------------------

__global__ void k_fill_f32(float* __restrict__ p, int n, float v) {
  int i = blockIdx.x * blockDim.x + threadIdx.x;
  if (i < n) p[i] = v;
}

// ne = emb[1000x20] @ memory[20x64] -> f16 [NP x 64], pad rows zeroed
__global__ void k_ne(const float* __restrict__ emb, const float* __restrict__ mem,
                     half_t* __restrict__ out) {
  int i = blockIdx.x * blockDim.x + threadIdx.x;
  if (i >= NP * 64) return;
  int d = i & 63, n = i >> 6;
  float s = 0.f;
  if (n < NN_) {
#pragma unroll
    for (int j = 0; j < 20; ++j) s += emb[n * 20 + j] * mem[j * 64 + d];
  }
  out[i] = (half_t)s;
}

// row softmax(relu(.)) of scores[NP x NP] -> G (f16) and its transpose GT
__global__ void k_relu_softmax_f16(const float* __restrict__ S,
                                   half_t* __restrict__ G,
                                   half_t* __restrict__ GT) {
  const int row = blockIdx.x;
  const int tid = threadIdx.x;
  __shared__ float red[256];
  if (row >= NN_) {
    for (int j = tid; j < NP; j += 256) {
      G[(size_t)row * NP + j]  = (half_t)0.f;
      GT[(size_t)j * NP + row] = (half_t)0.f;
    }
    return;
  }
  const float* Sr = S + (size_t)row * NP;
  float mx = 0.f;
  for (int j = tid; j < NN_; j += 256) {
    float v = Sr[j]; v = v > 0.f ? v : 0.f;
    mx = fmaxf(mx, v);
  }
  red[tid] = mx; __syncthreads();
  for (int s = 128; s > 0; s >>= 1) {
    if (tid < s) red[tid] = fmaxf(red[tid], red[tid + s]);
    __syncthreads();
  }
  mx = red[0]; __syncthreads();
  float sum = 0.f;
  for (int j = tid; j < NN_; j += 256) {
    float v = Sr[j]; v = v > 0.f ? v : 0.f;
    sum += __expf(v - mx);
  }
  red[tid] = sum; __syncthreads();
  for (int s = 128; s > 0; s >>= 1) {
    if (tid < s) red[tid] += red[tid + s];
    __syncthreads();
  }
  float inv = 1.f / red[0];
  for (int j = tid; j < NP; j += 256) {
    float v = 0.f;
    if (j < NN_) {
      float x = Sr[j]; x = x > 0.f ? x : 0.f;
      v = __expf(x - mx) * inv;
    }
    half_t hv = (half_t)v;
    G[(size_t)row * NP + j]  = hv;
    GT[(size_t)j * NP + row] = hv;
  }
}

// zero-pad + f16 + transpose weights: src[6*C x Dout] -> dst[6][Dout][Cp]
__global__ void k_pad_weight_t(const float* __restrict__ src, half_t* __restrict__ dst,
                               int C, int Cp, int Dout) {
  int i = blockIdx.x * blockDim.x + threadIdx.x;
  int total = 6 * Cp * Dout;
  if (i >= total) return;
  int cp = i % Cp;
  int d  = (i / Cp) % Dout;
  int k  = i / (Cp * Dout);
  float v = (cp < C) ? src[(size_t)(k * C + cp) * Dout + d] : 0.f;
  dst[i] = (half_t)v;
}

// Build concat input X[NP][B*Cp] (f16) and its transpose Xt[B*Cp][NP].
// value = [input(din) | state(dout) (*z) | 0pad]
// mode 0: srcA = x[B,T,N,2] gather (encoder layer0)
// mode 1: srcA = prev-layer state [NB x din]
// mode 2: c==0 -> go[row], c==1 -> y_cov[B,T,N,1] gather (decoder layer0)
__global__ void k_xcat(half_t* __restrict__ X, half_t* __restrict__ Xt,
                       int mode, int t,
                       const float* __restrict__ srcA, const float* __restrict__ srcB,
                       const float* __restrict__ state, const float* __restrict__ zr,
                       int din, int dout, int Cp) {
  int i = blockIdx.x * blockDim.x + threadIdx.x;
  int total = NP * BB * Cp;
  if (i >= total) return;
  int c = i % Cp;
  int b = (i / Cp) & 31;
  int n = i / (Cp * BB);
  float v = 0.f;
  if (n < NN_) {
    int row = n * BB + b;
    if (c < din) {
      if (mode == 0)       v = srcA[((size_t)b * TT + t) * (NN_ * 2) + n * 2 + c];
      else if (mode == 1)  v = srcA[(size_t)row * din + c];
      else                 v = (c == 0) ? srcA[row]
                                        : srcB[((size_t)b * TT + t) * NN_ + n];
    } else if (c < din + dout) {
      float s = state[(size_t)row * dout + (c - din)];
      if (zr) s *= zr[(size_t)row * (2 * dout) + (c - din)];  // z = zr[:, :dout]
      v = s;
    }
  }
  half_t hv = (half_t)v;
  X[i] = hv;
  Xt[(size_t)(b * Cp + c) * NP + n] = hv;
}

// zr = sigmoid(zr + gb)  over NB rows x D cols
__global__ void k_sigmoid_bias(float* __restrict__ zr, const float* __restrict__ gb,
                               int D, int total) {
  int i = blockIdx.x * blockDim.x + threadIdx.x;
  if (i >= total) return;
  int d = i % D;
  float v = zr[i] + gb[d];
  zr[i] = 1.f / (1.f + __expf(-v));
}

// state = r*state + (1-r)*tanh(hc + ub),  r = zr[:, dout:]
__global__ void k_combine(float* __restrict__ state, const float* __restrict__ zr,
                          const float* __restrict__ hc, const float* __restrict__ ub,
                          int dout, int total) {
  int i = blockIdx.x * blockDim.x + threadIdx.x;
  if (i >= total) return;
  int d = i % dout;
  int r = i / dout;
  float rg = zr[(size_t)r * (2 * dout) + dout + d];
  float h = tanhf(hc[i] + ub[d]);
  state[i] = rg * state[i] + (1.f - rg) * h;
}

// Memory network: query = h@Wq, att = softmax(query@mem^T), value = att@mem,
// decoder init states d0 = d1 = [h | value].  One 64-thread block per row.
__global__ void k_memnet(const float* __restrict__ h, const float* __restrict__ Wq,
                         const float* __restrict__ mem,
                         float* __restrict__ d0, float* __restrict__ d1) {
  int r = blockIdx.x;                 // 0..NB-1
  int tid = threadIdx.x;              // 64
  __shared__ float hv[64];
  __shared__ float q[64];
  __shared__ float sc[20];
  hv[tid] = h[(size_t)r * 64 + tid];
  __syncthreads();
  float s = 0.f;
#pragma unroll 8
  for (int e = 0; e < 64; ++e) s += hv[e] * Wq[e * 64 + tid];
  q[tid] = s;
  __syncthreads();
  if (tid < 20) {
    float t = 0.f;
#pragma unroll 8
    for (int d = 0; d < 64; ++d) t += q[d] * mem[tid * 64 + d];
    sc[tid] = t;
  }
  __syncthreads();
  float mx = sc[0];
#pragma unroll
  for (int j = 1; j < 20; ++j) mx = fmaxf(mx, sc[j]);
  float sum = 0.f;
#pragma unroll
  for (int j = 0; j < 20; ++j) sum += __expf(sc[j] - mx);
  float inv = 1.f / sum;
  float v = 0.f;
#pragma unroll
  for (int j = 0; j < 20; ++j) v += (__expf(sc[j] - mx) * inv) * mem[j * 64 + tid];
  d0[(size_t)r * 128 + tid]      = hv[tid];
  d0[(size_t)r * 128 + 64 + tid] = v;
  d1[(size_t)r * 128 + tid]      = hv[tid];
  d1[(size_t)r * 128 + 64 + tid] = v;
}

// go = st@proj_W + proj_b; also scatter to output[b][t][n][0]
__global__ void k_proj_out(const float* __restrict__ st, const float* __restrict__ W,
                           const float* __restrict__ bias, float* __restrict__ go,
                           float* __restrict__ out, int t) {
  int r = blockIdx.x * blockDim.x + threadIdx.x;
  if (r >= NB) return;
  float s = bias[0];
#pragma unroll 8
  for (int d = 0; d < 128; ++d) s += st[(size_t)r * 128 + d] * W[d];
  go[r] = s;
  int n = r >> 5, b = r & 31;
  out[((size_t)b * TT + t) * NN_ + n] = s;
}

// ---------------------------------------------------------------------------
// Host-side orchestration
// ---------------------------------------------------------------------------

extern "C" void kernel_launch(void* const* d_in, const int* in_sizes, int n_in,
                              void* d_out, int out_size, void* d_ws, size_t ws_size,
                              hipStream_t stream) {
  (void)in_sizes; (void)n_in; (void)out_size; (void)ws_size;
  const float* x      = (const float*)d_in[0];
  const float* ycov   = (const float*)d_in[1];
  const float* emb0   = (const float*)d_in[2];
  const float* emb1   = (const float*)d_in[3];
  const float* Wq     = (const float*)d_in[4];
  const float* memry  = (const float*)d_in[5];
  const float* gW[4] = {(const float*)d_in[6],  (const float*)d_in[10],
                        (const float*)d_in[14], (const float*)d_in[18]};
  const float* gb[4] = {(const float*)d_in[7],  (const float*)d_in[11],
                        (const float*)d_in[15], (const float*)d_in[19]};
  const float* uW[4] = {(const float*)d_in[8],  (const float*)d_in[12],
                        (const float*)d_in[16], (const float*)d_in[20]};
  const float* ub[4] = {(const float*)d_in[9],  (const float*)d_in[13],
                        (const float*)d_in[17], (const float*)d_in[21]};
  const float* projW = (const float*)d_in[22];
  const float* projb = (const float*)d_in[23];
  float* out = (float*)d_out;

  // layer geometry: enc0, enc1, dec0, dec1
  const int DIN[4]  = {2, 64, 2, 128};
  const int DOUT[4] = {64, 64, 128, 128};
  const int CPAD[4] = {96, 128, 160, 256};   // (din+dout) -> mult of 32

  // ---- workspace carve-up (256B aligned) ----
  char* w = (char*)d_ws;
  auto alloc = [&](size_t bytes) -> void* {
    void* p = (void*)w;
    w += (bytes + 255) & ~(size_t)255;
    return p;
  };
  half_t* Sg1 = (half_t*)alloc((size_t)NP * NP * 2);
  half_t* St1 = (half_t*)alloc((size_t)NP * NP * 2);
  half_t* Sg0 = (half_t*)alloc((size_t)NP * NP * 2);
  half_t* St0 = (half_t*)alloc((size_t)NP * NP * 2);
  half_t* SgT = (half_t*)alloc((size_t)NP * NP * 2);   // transpose scratch
  half_t* ne0h = (half_t*)alloc((size_t)NP * 64 * 2);
  half_t* ne1h = (half_t*)alloc((size_t)NP * 64 * 2);
  float*  scores = (float*)alloc((size_t)NP * NP * 4);
  half_t* X  = (half_t*)alloc((size_t)NP * BB * 256 * 2);
  half_t* Xt = (half_t*)alloc((size_t)NP * BB * 256 * 2);
  half_t* Y  = (half_t*)alloc((size_t)NP * BB * 256 * 2);
  float*  zrb = (float*)alloc((size_t)NBP * 256 * 4);
  float*  hcb = (float*)alloc((size_t)NBP * 128 * 4);
  float*  st_e0 = (float*)alloc((size_t)NBP * 64 * 4);
  float*  st_e1 = (float*)alloc((size_t)NBP * 64 * 4);
  float*  st_d0 = (float*)alloc((size_t)NBP * 128 * 4);
  float*  st_d1 = (float*)alloc((size_t)NBP * 128 * 4);
  float*  go = (float*)alloc((size_t)NBP * 4);
  half_t* gWt[4]; half_t* uWt[4];
  for (int l = 0; l < 4; ++l) {
    gWt[l] = (half_t*)alloc((size_t)6 * CPAD[l] * (2 * DOUT[l]) * 2);
    uWt[l] = (half_t*)alloc((size_t)6 * CPAD[l] * DOUT[l] * 2);
  }

  auto fill0 = [&](float* p, size_t n) {
    k_fill_f32<<<(int)((n + 255) / 256), 256, 0, stream>>>(p, (int)n, 0.f);
  };

  // ---- f16-pad (transposed) GRU weights ----
  for (int l = 0; l < 4; ++l) {
    int C = DIN[l] + DOUT[l];
    int tg = 6 * CPAD[l] * 2 * DOUT[l];
    k_pad_weight_t<<<(tg + 255) / 256, 256, 0, stream>>>(gW[l], gWt[l], C, CPAD[l], 2 * DOUT[l]);
    int tu = 6 * CPAD[l] * DOUT[l];
    k_pad_weight_t<<<(tu + 255) / 256, 256, 0, stream>>>(uW[l], uWt[l], C, CPAD[l], DOUT[l]);
  }

  // ---- supports: S = [I, g1, 2g1^2-I, I, g0, 2g0^2-I] ----
  {
    int t = NP * 64;
    k_ne<<<(t + 255) / 256, 256, 0, stream>>>(emb0, memry, ne0h);
    k_ne<<<(t + 255) / 256, 256, 0, stream>>>(emb1, memry, ne1h);
    int Mt = NP / 16, Nt = NP / 16;
    int blocks = (Mt * (Nt / 4) + 3) / 4;
    // g1 = softmax(relu(ne1 @ ne0^T));  T2 = 2*g1@g1 - I
    k_gemm4_nt_f32<<<blocks, 128, 0, stream>>>(ne1h, ne0h, scores, Mt, Nt, 64, 64, 64, NP);
    k_relu_softmax_f16<<<NP, 256, 0, stream>>>(scores, Sg1, SgT);
    k_gemm4_nt_f16<<<blocks, 128, 0, stream>>>(Sg1, SgT, St1, Mt, Nt, NP, NP, NP, NP, 1);
    // g0 = softmax(relu(ne0 @ ne1^T));  T2 = 2*g0@g0 - I
    k_gemm4_nt_f32<<<blocks, 128, 0, stream>>>(ne0h, ne1h, scores, Mt, Nt, 64, 64, 64, NP);
    k_relu_softmax_f16<<<NP, 256, 0, stream>>>(scores, Sg0, SgT);
    k_gemm4_nt_f16<<<blocks, 128, 0, stream>>>(Sg0, SgT, St0, Mt, Nt, NP, NP, NP, NP, 1);
  }
  const half_t* Ssup[4] = {Sg1, St1, Sg0, St0};  // supports k = 1,2,4,5
  const int     kidx[4] = {1, 2, 4, 5};

  // AGCN: outbuf[NB x Dout] = sum_k (S_k @ X) @ W_k   (k=0,3 are identity)
  auto agcn = [&](float* outbuf, int Dout, half_t* const* Wt, int layer, int Cp) {
    size_t n = (size_t)NBP * Dout;
    fill0(outbuf, n);
    int cols = BB * Cp;
    int MtP = NBP / 16, NtW = Dout / 16;
    int pblocks = (MtP * (NtW / 4) + 3) / 4;
    const half_t* Wb = Wt[layer];
    // identity supports (Y := X)
    k_gemm4_nt_accf32<<<pblocks, 128, 0, stream>>>(X, Wb + (size_t)0 * Cp * Dout,
                                                   outbuf, MtP, NtW, Cp, Cp, Cp, Dout);
    k_gemm4_nt_accf32<<<pblocks, 128, 0, stream>>>(X, Wb + (size_t)3 * Cp * Dout,
                                                   outbuf, MtP, NtW, Cp, Cp, Cp, Dout);
    int MtG = NP / 16, NtG = cols / 16;
    int gblocks = (MtG * (NtG / 4) + 3) / 4;
    for (int i = 0; i < 4; ++i) {
      k_gemm4_nt_f16<<<gblocks, 128, 0, stream>>>(Ssup[i], Xt, Y, MtG, NtG, NP,
                                                  NP, NP, cols, 0);
      k_gemm4_nt_accf32<<<pblocks, 128, 0, stream>>>(Y, Wb + (size_t)kidx[i] * Cp * Dout,
                                                     outbuf, MtP, NtW, Cp, Cp, Cp, Dout);
    }
  };

  // One AGCRN (GRU) cell step, state updated in place.
  auto cell = [&](int mode, int t, const float* inA, const float* inB, float* state,
                  int layer) {
    int din = DIN[layer], dout = DOUT[layer], Cp = CPAD[layer];
    int tx = NP * BB * Cp;
    k_xcat<<<(tx + 255) / 256, 256, 0, stream>>>(X, Xt, mode, t, inA, inB, state,
                                                 nullptr, din, dout, Cp);
    agcn(zrb, 2 * dout, gWt, layer, Cp);
    int na = NB * 2 * dout;
    k_sigmoid_bias<<<(na + 255) / 256, 256, 0, stream>>>(zrb, gb[layer], 2 * dout, na);
    k_xcat<<<(tx + 255) / 256, 256, 0, stream>>>(X, Xt, mode, t, inA, inB, state,
                                                 zrb, din, dout, Cp);
    agcn(hcb, dout, uWt, layer, Cp);
    int nc = NB * dout;
    k_combine<<<(nc + 255) / 256, 256, 0, stream>>>(state, zrb, hcb, ub[layer], dout, nc);
  };

  // ---- encoder (step-interleaved stacked scan) ----
  fill0(st_e0, (size_t)NBP * 64);
  fill0(st_e1, (size_t)NBP * 64);
  for (int t = 0; t < TT; ++t) {
    cell(0, t, x, nullptr, st_e0, 0);
    cell(1, t, st_e0, nullptr, st_e1, 1);
  }

  // ---- memory network -> decoder initial states ----
  k_memnet<<<NB, 64, 0, stream>>>(st_e1, Wq, memry, st_d0, st_d1);

  // ---- decoder (autoregressive) ----
  fill0(go, NBP);
  for (int t = 0; t < TT; ++t) {
    cell(2, t, go, ycov, st_d0, 2);
    cell(1, t, st_d0, nullptr, st_d1, 3);
    k_proj_out<<<(NB + 255) / 256, 256, 0, stream>>>(st_d1, projW, projb, go, out, t);
  }
}
